// MSDeform_OnOffset_v1_80023830659195
// MI455X (gfx1250) — matrix operand order repaired and checked by reading, not compiled, not run on hardware
//
#include <hip/hip_runtime.h>
#include <hip/hip_bf16.h>
#include <math.h>

// ---------------------------------------------------------------------------
// MS-Deformable attention w/ offset embeddings, fused for MI455X (gfx1250).
//   K0: convert the three weight matrices to f16 (L2-resident thereafter).
//   K1: value projection GEMM (WMMA f16->f32), output head-major [b][h][pos][32].
//   K2: fused offset-embed GEMM (16q x 4096 x 256, WMMA) + SiLU + per-(h,l,k)
//       regression + softmax + bilinear sampling -> out_heads [b][q][256].
//   K3: output GEMM (WMMA) -> d_out.
// ---------------------------------------------------------------------------

#define D_MODEL   256
#define N_HEADS   8
#define N_LEVELS  4
#define N_POINTS  4
#define OFF_DIM   32
#define D_HEAD    32
#define LEN_IN    5440
#define LEN_Q     5440
#define N_BATCH   4
#define M_TOTAL   (N_BATCH * LEN_Q)      // 21760 rows
#define M_TILES   (M_TOTAL / 16)         // 1360 (5440 = 340*16, tiles never cross batch)
#define Q_TILES_PER_BATCH (LEN_Q / 16)   // 340

typedef _Float16 v16h __attribute__((ext_vector_type(16)));
typedef _Float16 v8h  __attribute__((ext_vector_type(8)));
typedef float    v8f  __attribute__((ext_vector_type(8)));

// Full scheduling fence: force a load batch to be issued (distinct registers,
// all in flight) before the consuming WMMA chain -> partial loadcnt waits.
__device__ __forceinline__ void sched_fence() {
#if __has_builtin(__builtin_amdgcn_sched_barrier)
  __builtin_amdgcn_sched_barrier(0);
#endif
}

// ---- WMMA fragment loaders (layouts per CDNA5 ISA 7.12.2) -----------------
// A 16x32 f16: lane m = lane&15; halves 0..7 = K k0..k0+7, halves 8..15 =
// K k0+16..k0+23, with k0 = 8*(lane>>4). Two contiguous 16B chunks.
__device__ __forceinline__ v16h frag_a(const _Float16* row /* &tile[m][kb*32] */, int lane) {
  const int k0 = (lane >> 4) * 8;
  v8h lo = *(const v8h*)(row + k0);
  v8h hi = *(const v8h*)(row + k0 + 16);
  return __builtin_shufflevector(lo, hi, 0,1,2,3,4,5,6,7,8,9,10,11,12,13,14,15);
}

// B 32x16 f16 (K x N), B[k][n] = W[n][k] for the  X @ W.T  convention:
// lane col n = lane&15; halves 0..7 = Wrow[koff..koff+7], halves 8..15 =
// Wrow[koff+16..koff+23], koff=8*(lane>>4). Two contiguous 16B chunks.
__device__ __forceinline__ v16h frag_b(const _Float16* wrow /* W + n*256 + kb*32 */, int lane) {
  const int koff = (lane >> 4) * 8;
  v8h lo = *(const v8h*)(wrow + koff);
  v8h hi = *(const v8h*)(wrow + koff + 16);
  return __builtin_shufflevector(lo, hi, 0,1,2,3,4,5,6,7,8,9,10,11,12,13,14,15);
}

__device__ __forceinline__ v8f wmma_f16(v16h a, v16h b, v8f c) {
  return __builtin_amdgcn_wmma_f32_16x16x32_f16(false, a, false, b, (short)0, c, false, false);
}

// ---- K0: weight conversion ------------------------------------------------
__global__ void prep_weights(const float* __restrict__ Wo, const float* __restrict__ Wv,
                             const float* __restrict__ Ww,
                             _Float16* __restrict__ WoH, _Float16* __restrict__ WvH,
                             _Float16* __restrict__ WwH) {
  int i = blockIdx.x * blockDim.x + threadIdx.x;            // 1,048,576 threads
  if (i < 4096 * 256) WoH[i] = (_Float16)Wo[i];
  if (i < 256 * 256) { WvH[i] = (_Float16)Wv[i]; WwH[i] = (_Float16)Ww[i]; }
}

// ---- K1/K3: 256-K GEMM, C = A(f32) x W.T(f16) + bias ----------------------
// MODE 0: output permuted to head-major value layout [b][h][pos][32]
// MODE 1: output row-major [m][256]
template <int MODE>
__global__ __launch_bounds__(256, 1)
void gemm256(const float* __restrict__ A, const _Float16* __restrict__ W,
             const float* __restrict__ bias, float* __restrict__ out) {
  __shared__ __align__(16) _Float16 As[16][D_MODEL];        // 8KB, A tile (f32->f16)
  const int m0 = blockIdx.x * 16;
  const int wave = threadIdx.x >> 5;
  const int lane = threadIdx.x & 31;
  const int n0 = blockIdx.y * 128 + wave * 16;

  for (int i = threadIdx.x; i < 16 * D_MODEL; i += 256) {
    int r = i >> 8, c = i & 255;
    As[r][c] = (_Float16)A[(m0 + r) * D_MODEL + c];
  }
  __syncthreads();

  const int mrow = lane & 15;
  const int ncol = n0 + (lane & 15);
  const _Float16* wrow = W + ncol * D_MODEL;

  // A fragments resident in registers (8 x 8 VGPRs)
  v16h a[8];
#pragma unroll
  for (int kb = 0; kb < 8; ++kb) a[kb] = frag_a(&As[mrow][kb * 32], lane);

  v8f acc;
#pragma unroll
  for (int i = 0; i < 8; ++i) acc[i] = 0.0f;

  // two batches of 4 B fragments; fence keeps each batch's 8 loads in flight
#pragma unroll
  for (int half = 0; half < 2; ++half) {
    v16h bf[4];
#pragma unroll
    for (int j = 0; j < 4; ++j) bf[j] = frag_b(wrow + (half * 4 + j) * 32, lane);
    sched_fence();
#pragma unroll
    for (int j = 0; j < 4; ++j) acc = wmma_f16(a[half * 4 + j], bf[j], acc);
  }

  const float bv = bias[ncol];
  const int rbase = m0 + 8 * (lane >> 4);                   // C/D: row = v + 8*(lane>=16)
#pragma unroll
  for (int v = 0; v < 8; ++v) {
    float r = acc[v] + bv;
    int m = rbase + v;
    if (MODE == 0) {
      int b  = m / LEN_IN, lin = m % LEN_IN;
      int nh = ncol >> 5,  nd  = ncol & 31;
      out[(((b * N_HEADS + nh) * LEN_IN) + lin) * D_HEAD + nd] = r;
    } else {
      out[m * D_MODEL + ncol] = r;
    }
  }
}

// ---- K2: fused offset GEMM + SiLU + regression + softmax + sampling -------
__device__ __forceinline__ float fetchv(const float* vb, int base, int size, int xi, int yi) {
  if ((unsigned)xi < (unsigned)size && (unsigned)yi < (unsigned)size)
    return vb[(size_t)(base + yi * size + xi) * D_HEAD];
  return 0.0f;
}

__global__ __launch_bounds__(256, 1)
void msdeform_core(const float* __restrict__ query, const float* __restrict__ refp,
                   const _Float16* __restrict__ WoH,
                   const float* __restrict__ Ws,  const float* __restrict__ Wsb,
                   const float* __restrict__ Wa,  const float* __restrict__ Wab,
                   const float* __restrict__ val, float* __restrict__ out_heads) {
  __shared__ __align__(16) _Float16 Qs[16][D_MODEL];        // 8KB query tile (f16)
  __shared__ float oe[N_HEADS][16][33];                     // ~16.9KB SiLU'd oe chunk, padded
  __shared__ float locs[16][N_HEADS][16][2];                // 16KB sampling locations
  __shared__ float aws[16][N_HEADS][16];                    // 8KB attn logits -> weights

  const int tile = blockIdx.x;                              // 0..1359
  const int b    = tile / Q_TILES_PER_BATCH;
  const int q0   = (tile % Q_TILES_PER_BATCH) * 16;
  const int wave = threadIdx.x >> 5;                        // wave == head
  const int lane = threadIdx.x & 31;
  const int h    = wave;

  // stage 16x256 query tile, f32 -> f16
  for (int i = threadIdx.x; i < 16 * D_MODEL; i += 256) {
    int r = i >> 8, c = i & 255;
    Qs[r][c] = (_Float16)query[(size_t)(b * LEN_Q + q0 + r) * D_MODEL + c];
  }
  __syncthreads();

  const int mrow = lane & 15;
  // A fragments (query tile) live in registers across all 16 groups
  v16h a[8];
#pragma unroll
  for (int kb = 0; kb < 8; ++kb) a[kb] = frag_a(&Qs[mrow][kb * 32], lane);

  // Phase 1: per (l,k) group of 32 oe columns: WMMA -> SiLU -> tiny regressions
  for (int lk = 0; lk < 16; ++lk) {
    const int g = h * 16 + lk;                              // column group, cols g*32..g*32+31
    const int l = lk >> 2;
    const _Float16* w0 = WoH + (size_t)(g * 32 +      (lane & 15)) * D_MODEL;
    const _Float16* w1 = WoH + (size_t)(g * 32 + 16 + (lane & 15)) * D_MODEL;

    // prefetch next group's weight rows toward near caches (global_prefetch_b8)
    if (lk < 15) {
      __builtin_prefetch(w0 + 32 * D_MODEL, 0, 3);
      __builtin_prefetch(w1 + 32 * D_MODEL, 0, 3);
    }

    v8f acc0, acc1;
#pragma unroll
    for (int i = 0; i < 8; ++i) { acc0[i] = 0.0f; acc1[i] = 0.0f; }

    // 4 batches of 4 B fragments (8 b128 loads each, kept in flight by the
    // sched fence); next batch's loads float up among the current WMMA chain.
#pragma unroll
    for (int half = 0; half < 2; ++half) {
      {
        v16h bf[4];
#pragma unroll
        for (int j = 0; j < 4; ++j) bf[j] = frag_b(w0 + (half * 4 + j) * 32, lane);
        sched_fence();
#pragma unroll
        for (int j = 0; j < 4; ++j) acc0 = wmma_f16(a[half * 4 + j], bf[j], acc0);
      }
      {
        v16h bf[4];
#pragma unroll
        for (int j = 0; j < 4; ++j) bf[j] = frag_b(w1 + (half * 4 + j) * 32, lane);
        sched_fence();
#pragma unroll
        for (int j = 0; j < 4; ++j) acc1 = wmma_f16(a[half * 4 + j], bf[j], acc1);
      }
    }

    // SiLU, park 16x32 chunk in LDS (wave-private slice; LDS is in-order per wave)
    const int row = 8 * (lane >> 4);
    const int col = lane & 15;
#pragma unroll
    for (int v = 0; v < 8; ++v) {
      float x0 = acc0[v], x1 = acc1[v];
      oe[wave][row + v][col]      = x0 / (1.0f + __expf(-x0));
      oe[wave][row + v][col + 16] = x1 / (1.0f + __expf(-x1));
    }
    // 32->2 offset and 32->1 attn regressions: lanes 0..15 -> (q, x + attn),
    // lanes 16..31 -> (q, y)
    const int q  = lane & 15;
    const int cc = lane >> 4;
    const float* wsrow = Ws + (size_t)g * OFF_DIM * 2;      // (H,L,K,32,2)
    const float* warow = Wa + (size_t)g * OFF_DIM;          // (H,L,K,32,1)
    float s = 0.0f, sa = 0.0f;
#pragma unroll
    for (int d = 0; d < OFF_DIM; ++d) {
      float o = oe[wave][q][d];
      s  += o * wsrow[d * 2 + cc];
      sa += o * warow[d];
    }
    const float norm = (float)(64 >> l);                    // square levels: W_l == H_l
    const float ref  = refp[(size_t)((b * LEN_Q + q0 + q) * N_LEVELS + l) * 2 + cc];
    locs[q][h][lk][cc] = ref + (s + Wsb[g * 2 + cc]) / norm;
    if (cc == 0) aws[q][h][lk] = sa + Wab[g];
  }
  __syncthreads();

  // Phase 2: softmax over the 16 (l,k) logits for each (q, head)
  if (threadIdx.x < 128) {
    int q = threadIdx.x >> 3, hh = threadIdx.x & 7;
    float mx = -1e30f;
#pragma unroll
    for (int i = 0; i < 16; ++i) mx = fmaxf(mx, aws[q][hh][i]);
    float e[16], sum = 0.0f;
#pragma unroll
    for (int i = 0; i < 16; ++i) { e[i] = __expf(aws[q][hh][i] - mx); sum += e[i]; }
    float inv = 1.0f / sum;
#pragma unroll
    for (int i = 0; i < 16; ++i) aws[q][hh][i] = e[i] * inv;
  }
  __syncthreads();

  // Phase 3: bilinear sampling. wave == head, lane == channel; every corner
  // gather is a coalesced 128B line from head-major value.
  const int d = lane;
  const float* vb = val + ((size_t)(b * N_HEADS + h)) * LEN_IN * D_HEAD + d;
  for (int q = 0; q < 16; ++q) {
    float accf = 0.0f;
#pragma unroll
    for (int lk = 0; lk < 16; ++lk) {
      const int l = lk >> 2;
      const int size = 64 >> l;
      const int base = (l == 0) ? 0 : (l == 1) ? 4096 : (l == 2) ? 5120 : 5376;
      float lx = locs[q][h][lk][0], ly = locs[q][h][lk][1];
      float awv = aws[q][h][lk];
      float x = lx * (float)size - 0.5f, y = ly * (float)size - 0.5f;
      float xf = floorf(x), yf = floorf(y);
      int   x0 = (int)xf,   y0 = (int)yf;
      float wx = x - xf,    wy = y - yf;
      float g00 = fetchv(vb, base, size, x0,     y0);
      float g01 = fetchv(vb, base, size, x0 + 1, y0);
      float g10 = fetchv(vb, base, size, x0,     y0 + 1);
      float g11 = fetchv(vb, base, size, x0 + 1, y0 + 1);
      float top = g00 * (1.0f - wx) + g01 * wx;
      float bot = g10 * (1.0f - wx) + g11 * wx;
      accf += awv * (top * (1.0f - wy) + bot * wy);
    }
    out_heads[(size_t)(b * LEN_Q + q0 + q) * D_MODEL + h * D_HEAD + d] = accf;
  }
}

// ---------------------------------------------------------------------------
extern "C" void kernel_launch(void* const* d_in, const int* in_sizes, int n_in,
                              void* d_out, int out_size, void* d_ws, size_t ws_size,
                              hipStream_t stream) {
  (void)in_sizes; (void)n_in; (void)out_size; (void)ws_size;
  const float* query = (const float*)d_in[0];
  const float* refp  = (const float*)d_in[1];
  const float* xin   = (const float*)d_in[2];
  // d_in[3] = input_spatial_shapes (static pyramid, unused)
  const float* Wo    = (const float*)d_in[4];
  const float* Ws    = (const float*)d_in[5];
  const float* Wsb   = (const float*)d_in[6];
  const float* Wa    = (const float*)d_in[7];
  const float* Wab   = (const float*)d_in[8];
  const float* Wv    = (const float*)d_in[9];
  const float* bv    = (const float*)d_in[10];
  const float* Ww    = (const float*)d_in[11];
  const float* bw    = (const float*)d_in[12];
  float* out = (float*)d_out;

  // workspace layout (bytes)
  char* ws = (char*)d_ws;
  _Float16* WoH  = (_Float16*)(ws);                         // 4096*256*2  = 2,097,152
  _Float16* WvH  = (_Float16*)(ws + 2097152);               //  256*256*2  =   131,072
  _Float16* WwH  = (_Float16*)(ws + 2228224);               //  256*256*2  =   131,072
  float*    val  = (float*)   (ws + 2359296);               // 4*8*5440*32*4 = 22,282,240
  float*    outh = (float*)   (ws + 24641536);              // 4*5440*256*4  = 22,282,240
                                                            // total ~44.8 MB

  prep_weights<<<4096, 256, 0, stream>>>(Wo, Wv, Ww, WoH, WvH, WwH);
  gemm256<0><<<dim3(M_TILES, 2), 256, 0, stream>>>(xin, WvH, bv, val);
  msdeform_core<<<M_TILES, 256, 0, stream>>>(query, refp, WoH, Ws, Wsb, Wa, Wab, val, outh);
  gemm256<1><<<dim3(M_TILES, 2), 256, 0, stream>>>(outh, WwH, bw, out);
}
